// TransformerDecoderBlock_14611478741286
// MI455X (gfx1250) — matrix operand order repaired and checked by reading, not compiled, not run on hardware
//
#include <hip/hip_runtime.h>

// ---------------------------------------------------------------------------
// MI455X (gfx1250) transformer decoder block.
// bf16 WMMA everywhere, fp32 accumulate. GEMM: block-cooperative LDS-staged
// B tiles, double-buffered via Tensor Data Mover (tensor_load_to_lds +
// s_wait_tensorcnt), A fragments register-prefetched, B fragments loaded as
// one LDS clause then 4 back-to-back WMMAs. Flash attention with online
// softmax and LDS C->A fragment re-layout.
// ---------------------------------------------------------------------------

typedef __bf16 bf16;
typedef __attribute__((ext_vector_type(16))) __bf16 v16bf;
typedef __attribute__((ext_vector_type(8)))  __bf16 v8bf;
typedef __attribute__((ext_vector_type(8)))  float  v8f;
typedef unsigned int u32x4 __attribute__((ext_vector_type(4)));
typedef int i32x4 __attribute__((ext_vector_type(4)));
typedef int i32x8v __attribute__((ext_vector_type(8)));

#define BB 4
#define SS 1024
#define DDIM 1024
#define HH 16
#define DH 64
#define FF 4096
#define MROWS (BB * SS)   // 4096

#if defined(__gfx1250__) && defined(__has_builtin)
#if __has_builtin(__builtin_amdgcn_tensor_load_to_lds)
#define HAVE_TDM 1
#endif
#endif
#ifndef HAVE_TDM
#define HAVE_TDM 0
#endif

union FragA { v16bf v; v8bf h[2]; };

__device__ __forceinline__ bf16 f2bf(float f) {
  union { float f; unsigned u; } a; a.f = f;
  unsigned u = a.u;
  u += 0x7FFFu + ((u >> 16) & 1u);            // round-to-nearest-even
  unsigned short s = (unsigned short)(u >> 16);
  union { unsigned short s; bf16 b; } c; c.s = s;
  return c.b;
}

#if HAVE_TDM
// Issue a TDM load of a 64-row x 32-elem bf16 tile (row stride K elems) to LDS.
__device__ __forceinline__ void tdm_load_tile(const bf16* gsrc, unsigned lds_off,
                                              int K, int N) {
  unsigned long long ga = (unsigned long long)(uintptr_t)gsrc;
  u32x4 g0;
  g0[0] = 1u;                                                  // count=1 (valid D#)
  g0[1] = lds_off;                                             // LDS byte address
  g0[2] = (unsigned)ga;                                        // global addr [31:0]
  g0[3] = (unsigned)((ga >> 32) & 0x01FFFFFFu) | (2u << 30);   // addr[56:32] + type=2
  i32x8v g1;
  g1[0] = 1 << 16;                                             // data_size=1 (2 bytes)
  g1[1] = (int)(((unsigned)K & 0xFFFFu) << 16);                // tensor_dim0[15:0]
  g1[2] = (int)((((unsigned)K >> 16) & 0xFFFFu) |
                (((unsigned)N & 0xFFFFu) << 16));              // td0 hi | tensor_dim1 lo
  g1[3] = (int)((((unsigned)N >> 16) & 0xFFFFu) | (32u << 16));// td1 hi | tile_dim0=32
  g1[4] = 64;                                                  // tile_dim1=64, tile_dim2=0
  g1[5] = K;                                                   // tensor_dim0_stride lo
  g1[6] = 0;                                                   // stride hi / dim1_stride lo
  g1[7] = 0;
  i32x4 z4 = {0, 0, 0, 0};
#if __clang_major__ >= 23
  i32x8v z8 = {0, 0, 0, 0, 0, 0, 0, 0};
  __builtin_amdgcn_tensor_load_to_lds(g0, g1, z4, z4, z8, 0);
#else
  __builtin_amdgcn_tensor_load_to_lds(g0, g1, z4, z4, 0);
#endif
}
#endif

// ---------------------------------------------------------------------------
// fp32 -> bf16 elementwise
// ---------------------------------------------------------------------------
__global__ void __launch_bounds__(256) cvt_bf16(const float* __restrict__ x,
                                                bf16* __restrict__ y, int n) {
  int i = blockIdx.x * 256 + threadIdx.x;
  if (i < n) y[i] = f2bf(x[i]);
}

// fp32 W[K,N] row-major -> bf16 Wt[N,K] row-major (B-operand friendly)
__global__ void __launch_bounds__(256) cvt_w_t(const float* __restrict__ W,
                                               bf16* __restrict__ Wt, int K, int N) {
  int i = blockIdx.x * 256 + threadIdx.x;
  if (i >= N * K) return;
  int n = i / K, k = i - n * K;
  Wt[i] = f2bf(W[(size_t)k * N + n]);
}

// ---------------------------------------------------------------------------
// GEMM: C[M,N] = A[M,K](bf16) x Wt[N,K](bf16)^T + bias (+ resid)(+relu)
// Block: 8 waves over 128 rows x 64 cols. B tile (64x32 bf16) staged in LDS
// via TDM, double buffered, reused by all 8 waves. A fragments prefetched.
// ---------------------------------------------------------------------------
__global__ void __launch_bounds__(256) gemm_bf16_wmma(
    const bf16* __restrict__ A, const bf16* __restrict__ Wt,
    const float* __restrict__ bias, const float* __restrict__ resid,
    float* __restrict__ outF, bf16* __restrict__ outB,
    int M, int N, int K, int relu, int vstoreT) {
  __shared__ __align__(16) bf16 Bt[2][64 * 32];

  const int tid  = threadIdx.x;
  const int lane = tid & 31;
  const int half = lane >> 4;
  const int l16  = lane & 15;
  const int wib  = tid >> 5;

  const int nbn = N >> 6;
  const int bm  = blockIdx.x / nbn;
  const int bn  = blockIdx.x - bm * nbn;
  const int m0  = bm * 128 + wib * 16;
  const int n0  = bn * 64;

  const bf16* Arow = A + (size_t)(m0 + l16) * K + half * 8;
  const bf16* Bsrc = Wt + (size_t)n0 * K;

  // ---- stage k=0 B tile into buffer 0 ----
#if HAVE_TDM
  if (wib == 0) tdm_load_tile(Bsrc, (unsigned)(uintptr_t)&Bt[0][0], K, N);
#else
  {
    const int row = tid >> 2, c = (tid & 3) * 8;
    *(v8bf*)(&Bt[0][row * 32 + c]) = *(const v8bf*)(Bsrc + (size_t)row * K + c);
  }
#endif

  FragA aCur;
  aCur.h[0] = *(const v8bf*)(Arow);
  aCur.h[1] = *(const v8bf*)(Arow + 16);

  v8f acc[4] = {};
  int cur = 0;
  for (int k0 = 0; k0 < K; k0 += 32, cur ^= 1) {
    const int nxt = cur ^ 1;
    const bool hn = (k0 + 32) < K;
    __syncthreads();                  // prior readers of buf[nxt] are done
#if HAVE_TDM
    if (wib == 0) {
      if (hn) {
        tdm_load_tile(Bsrc + k0 + 32, (unsigned)(uintptr_t)&Bt[nxt][0], K, N);
        __builtin_amdgcn_s_wait_tensorcnt(1);   // previous tile (buf[cur]) done
      } else {
        __builtin_amdgcn_s_wait_tensorcnt(0);   // last tile done
      }
    }
#else
    if (hn) {
      const int row = tid >> 2, c = (tid & 3) * 8;
      *(v8bf*)(&Bt[nxt][row * 32 + c]) =
          *(const v8bf*)(Bsrc + (size_t)row * K + k0 + 32 + c);
    }
#endif
    __syncthreads();                  // buf[cur] is ready for everyone

    // prefetch next A fragment (global load overlaps the WMMAs below)
    FragA aN;
    const bf16* An = Arow + (hn ? (k0 + 32) : k0);
    aN.h[0] = *(const v8bf*)(An);
    aN.h[1] = *(const v8bf*)(An + 16);

    // load all 4 B fragments as one LDS clause, then 4 back-to-back WMMAs
    FragA bfr[4];
#pragma unroll
    for (int t = 0; t < 4; ++t) {
      const bf16* Br = &Bt[cur][(t * 16 + l16) * 32 + half * 8];
      bfr[t].h[0] = *(const v8bf*)(Br);
      bfr[t].h[1] = *(const v8bf*)(Br + 16);
    }
#pragma unroll
    for (int t = 0; t < 4; ++t) {
      acc[t] = __builtin_amdgcn_wmma_f32_16x16x32_bf16(
          false, aCur.v, false, bfr[t].v, (short)0, acc[t], false, false);
    }
    aCur = aN;
  }

  // ---- epilogue ----
#pragma unroll
  for (int t = 0; t < 4; ++t) {
#pragma unroll
    for (int r = 0; r < 8; ++r) {
      const int row = m0 + r + half * 8;
      const int col = n0 + t * 16 + l16;
      float v = acc[t][r];
      if (bias)  v += bias[col];
      if (resid) v += resid[(size_t)row * N + col];
      if (relu)  v = v > 0.f ? v : 0.f;
      if (outF)  outF[(size_t)row * N + col] = v;
      if (outB) {
        size_t idx;
        if (vstoreT) {               // V projection: store as Vt[B,H,DH,S]
          const int bb = row >> 10, s = row & (SS - 1);
          const int hh = col >> 6,  d = col & (DH - 1);
          idx = ((((size_t)bb * HH + hh) * DH + d) << 10) | (unsigned)s;
        } else {
          idx = (size_t)row * N + col;
        }
        outB[idx] = f2bf(v);
      }
    }
  }
}

// ---------------------------------------------------------------------------
// Flash attention: one wave per (b, h, 16-row query tile).
// Q,K: bf16 [B,S,D] (head-contiguous). Vt: bf16 [B,H,DH,S].
// Out: bf16 [B,S,D]. causal=1 -> causal mask; else key < vlens[b].
// ---------------------------------------------------------------------------
__global__ void __launch_bounds__(256) attn_wmma(
    const bf16* __restrict__ Q, const bf16* __restrict__ Kmat,
    const bf16* __restrict__ Vt, bf16* __restrict__ Out,
    const int* __restrict__ vlens, int causal) {
  __shared__ bf16 plds[8][16 * 32];       // per-wave 16x32 P tile (bf16)

  const int lane = threadIdx.x & 31;
  const int half = lane >> 4;
  const int l16  = lane & 15;
  const int wib  = threadIdx.x >> 5;
  const int gw   = blockIdx.x * 8 + wib;
  const int qt = gw & 63;
  const int h  = (gw >> 6) & (HH - 1);
  const int b  = gw >> 10;
  const int q0 = qt << 4;

  // Q fragments (d 0..31 and 32..63), loaded once
  const bf16* Qrow = Q + ((size_t)(b * SS + q0 + l16)) * DDIM + h * DH + half * 8;
  FragA aQ0, aQ1;
  aQ0.h[0] = *(const v8bf*)(Qrow);
  aQ0.h[1] = *(const v8bf*)(Qrow + 16);
  aQ1.h[0] = *(const v8bf*)(Qrow + 32);
  aQ1.h[1] = *(const v8bf*)(Qrow + 48);

  v8f Oc[4] = {};
  float mrow[8], lrow[8];
#pragma unroll
  for (int r = 0; r < 8; ++r) { mrow[r] = -1e30f; lrow[r] = 0.f; }

  const int kmax = causal ? (q0 + 16) : vlens[b];

  for (int k0 = 0; k0 < kmax; k0 += 32) {
    // ---- scores: load all 4 K fragments first, then 4 WMMAs ----
    FragA kb[4];   // [0,1]: keys k0..+15 (d 0-31, 32-63); [2,3]: keys k0+16..+31
    {
      const bf16* Kr = Kmat + ((size_t)(b * SS + k0 + l16)) * DDIM + h * DH + half * 8;
      kb[0].h[0] = *(const v8bf*)(Kr);      kb[0].h[1] = *(const v8bf*)(Kr + 16);
      kb[1].h[0] = *(const v8bf*)(Kr + 32); kb[1].h[1] = *(const v8bf*)(Kr + 48);
    }
    {
      const bf16* Kr = Kmat + ((size_t)(b * SS + k0 + 16 + l16)) * DDIM + h * DH + half * 8;
      kb[2].h[0] = *(const v8bf*)(Kr);      kb[2].h[1] = *(const v8bf*)(Kr + 16);
      kb[3].h[0] = *(const v8bf*)(Kr + 32); kb[3].h[1] = *(const v8bf*)(Kr + 48);
    }
    v8f sc0 = {}, sc1 = {};
    sc0 = __builtin_amdgcn_wmma_f32_16x16x32_bf16(false, aQ0.v, false, kb[0].v, (short)0, sc0, false, false);
    sc0 = __builtin_amdgcn_wmma_f32_16x16x32_bf16(false, aQ1.v, false, kb[1].v, (short)0, sc0, false, false);
    sc1 = __builtin_amdgcn_wmma_f32_16x16x32_bf16(false, aQ0.v, false, kb[2].v, (short)0, sc1, false, false);
    sc1 = __builtin_amdgcn_wmma_f32_16x16x32_bf16(false, aQ1.v, false, kb[3].v, (short)0, sc1, false, false);

    // ---- mask + online softmax (row lives in one VGPR across a half-wave) ----
    float p0[8], p1[8];
#pragma unroll
    for (int r = 0; r < 8; ++r) {
      const int row = q0 + r + half * 8;
      const int c0 = k0 + l16;
      const int c1 = k0 + 16 + l16;
      float v0 = sc0[r] * 0.125f;                 // 1/sqrt(64)
      float v1 = sc1[r] * 0.125f;
      const bool ok0 = causal ? (c0 <= row) : (c0 < kmax);
      const bool ok1 = causal ? (c1 <= row) : (c1 < kmax);
      v0 = ok0 ? v0 : -1.0e6f;
      v1 = ok1 ? v1 : -1.0e6f;
      float mx = fmaxf(v0, v1);
#pragma unroll
      for (int o = 1; o < 16; o <<= 1) mx = fmaxf(mx, __shfl_xor(mx, o, 32));
      const float mnew  = fmaxf(mrow[r], mx);
      const float alpha = __expf(mrow[r] - mnew);
      const float e0 = __expf(v0 - mnew);
      const float e1 = __expf(v1 - mnew);
      float ps = e0 + e1;
#pragma unroll
      for (int o = 1; o < 16; o <<= 1) ps += __shfl_xor(ps, o, 32);
      lrow[r] = lrow[r] * alpha + ps;
      mrow[r] = mnew;
#pragma unroll
      for (int t = 0; t < 4; ++t) Oc[t][r] *= alpha;
      p0[r] = e0; p1[r] = e1;
    }

    // ---- re-layout P: C-fragment -> LDS row-major -> A-fragment ----
    bf16* pl = plds[wib];
#pragma unroll
    for (int r = 0; r < 8; ++r) {
      const int rr = r + half * 8;
      pl[rr * 32 + l16]      = f2bf(p0[r]);
      pl[rr * 32 + 16 + l16] = f2bf(p1[r]);
    }
    asm volatile("s_wait_dscnt 0" ::: "memory");
    FragA aP;
    const bf16* pr = pl + (size_t)l16 * 32 + half * 8;
    aP.h[0] = *(const v8bf*)(pr);
    aP.h[1] = *(const v8bf*)(pr + 16);

    // ---- O += P(16x32) x V(32x64): load all 4 V fragments, then 4 WMMAs ----
    FragA bv[4];
#pragma unroll
    for (int t = 0; t < 4; ++t) {
      const bf16* Vr = Vt + (((size_t)b * HH + h) * DH + t * 16 + l16) * SS + k0 + half * 8;
      bv[t].h[0] = *(const v8bf*)(Vr);
      bv[t].h[1] = *(const v8bf*)(Vr + 16);
    }
#pragma unroll
    for (int t = 0; t < 4; ++t) {
      Oc[t] = __builtin_amdgcn_wmma_f32_16x16x32_bf16(false, aP.v, false, bv[t].v, (short)0, Oc[t], false, false);
    }
  }

  // ---- normalize + store bf16 [B,S,D] ----
#pragma unroll
  for (int r = 0; r < 8; ++r) {
    const float inv = 1.f / fmaxf(lrow[r], 1e-20f);
    const int row = q0 + r + half * 8;
#pragma unroll
    for (int t = 0; t < 4; ++t) {
      Out[((size_t)(b * SS + row)) * DDIM + h * DH + t * 16 + l16] = f2bf(Oc[t][r] * inv);
    }
  }
}

// ---------------------------------------------------------------------------
// LayerNorm over last dim D; writes fp32 and optional bf16 copy.
// ---------------------------------------------------------------------------
__global__ void __launch_bounds__(256) ln_kernel(
    const float* __restrict__ x, const float* __restrict__ g,
    const float* __restrict__ be, float* __restrict__ yF,
    bf16* __restrict__ yB, int D) {
  const int row = blockIdx.x;
  const float* xr = x + (size_t)row * D;
  const int tid = threadIdx.x;
  const int lane = tid & 31, wib = tid >> 5;
  __shared__ float red[8];

  float s = 0.f;
  for (int i = tid; i < D; i += 256) s += xr[i];
#pragma unroll
  for (int o = 1; o < 32; o <<= 1) s += __shfl_xor(s, o, 32);
  if (lane == 0) red[wib] = s;
  __syncthreads();
  float tot = 0.f;
#pragma unroll
  for (int w = 0; w < 8; ++w) tot += red[w];
  const float mean = tot / (float)D;
  __syncthreads();

  float v = 0.f;
  for (int i = tid; i < D; i += 256) { float d = xr[i] - mean; v += d * d; }
#pragma unroll
  for (int o = 1; o < 32; o <<= 1) v += __shfl_xor(v, o, 32);
  if (lane == 0) red[wib] = v;
  __syncthreads();
  float vt = 0.f;
#pragma unroll
  for (int w = 0; w < 8; ++w) vt += red[w];
  const float rstd = rsqrtf(vt / (float)D + 1e-5f);

  for (int i = tid; i < D; i += 256) {
    const float o = (xr[i] - mean) * rstd * g[i] + be[i];
    if (yF) yF[(size_t)row * D + i] = o;
    if (yB) yB[(size_t)row * D + i] = f2bf(o);
  }
}

// ---------------------------------------------------------------------------
extern "C" void kernel_launch(void* const* d_in, const int* in_sizes, int n_in,
                              void* d_out, int out_size, void* d_ws, size_t ws_size,
                              hipStream_t stream) {
  (void)in_sizes; (void)n_in; (void)out_size; (void)ws_size;

  const float* X   = (const float*)d_in[0];
  const float* ENC = (const float*)d_in[1];
  const int*   VL  = (const int*)d_in[2];
  const float* Wq1 = (const float*)d_in[3];  const float* bq1 = (const float*)d_in[4];
  const float* Wk1 = (const float*)d_in[5];  const float* bk1 = (const float*)d_in[6];
  const float* Wv1 = (const float*)d_in[7];  const float* bv1 = (const float*)d_in[8];
  const float* Wo1 = (const float*)d_in[9];  const float* bo1 = (const float*)d_in[10];
  const float* g1  = (const float*)d_in[11]; const float* be1 = (const float*)d_in[12];
  const float* Wq2 = (const float*)d_in[13]; const float* bq2 = (const float*)d_in[14];
  const float* Wk2 = (const float*)d_in[15]; const float* bk2 = (const float*)d_in[16];
  const float* Wv2 = (const float*)d_in[17]; const float* bv2 = (const float*)d_in[18];
  const float* Wo2 = (const float*)d_in[19]; const float* bo2 = (const float*)d_in[20];
  const float* g2  = (const float*)d_in[21]; const float* be2 = (const float*)d_in[22];
  const float* Wf1 = (const float*)d_in[23]; const float* bf1 = (const float*)d_in[24];
  const float* Wf2 = (const float*)d_in[25]; const float* bf2 = (const float*)d_in[26];
  const float* g3  = (const float*)d_in[27]; const float* be3 = (const float*)d_in[28];
  float* OUT = (float*)d_out;

  char* ws = (char*)d_ws;
  size_t off = 0;
  auto alloc = [&](size_t bytes) -> void* {
    void* p = ws + off;
    off += (bytes + 255) & ~(size_t)255;
    return p;
  };

  const size_t M = MROWS, D = DDIM, F = FF;
  // weights (bf16, transposed [N,K])
  bf16* Wq1t = (bf16*)alloc(D * D * 2);
  bf16* Wk1t = (bf16*)alloc(D * D * 2);
  bf16* Wv1t = (bf16*)alloc(D * D * 2);
  bf16* Wo1t = (bf16*)alloc(D * D * 2);
  bf16* Wq2t = (bf16*)alloc(D * D * 2);
  bf16* Wk2t = (bf16*)alloc(D * D * 2);
  bf16* Wv2t = (bf16*)alloc(D * D * 2);
  bf16* Wo2t = (bf16*)alloc(D * D * 2);
  bf16* Wf1t = (bf16*)alloc(D * F * 2);
  bf16* Wf2t = (bf16*)alloc(F * D * 2);
  // activations
  bf16* Xb  = (bf16*)alloc(M * D * 2);
  bf16* Eb  = (bf16*)alloc(M * D * 2);
  bf16* Qb  = (bf16*)alloc(M * D * 2);   // reused by both attentions
  bf16* Kb  = (bf16*)alloc(M * D * 2);
  bf16* Vtb = (bf16*)alloc(M * D * 2);
  bf16* Atn = (bf16*)alloc(M * D * 2);
  float* Tmp = (float*)alloc(M * D * 4); // reused pre-LN buffer
  float* Yf  = (float*)alloc(M * D * 4);
  bf16*  Yb  = (bf16*)alloc(M * D * 2);
  float* Zf  = (float*)alloc(M * D * 4);
  bf16*  Zb  = (bf16*)alloc(M * D * 2);
  bf16*  Hb  = (bf16*)alloc(M * F * 2);

  dim3 blk(256);
  auto cvtw = [&](const float* W, bf16* Wt, int K, int N) {
    int n = K * N;
    cvt_w_t<<<dim3((n + 255) / 256), blk, 0, stream>>>(W, Wt, K, N);
  };
  auto gemm = [&](const bf16* A_, const bf16* Wt_, const float* bias_,
                  const float* res_, float* oF, bf16* oB,
                  int Mm, int Nn, int Kk, int relu, int vst) {
    int blocks = (Mm / 128) * (Nn / 64);
    gemm_bf16_wmma<<<dim3(blocks), blk, 0, stream>>>(
        A_, Wt_, bias_, res_, oF, oB, Mm, Nn, Kk, relu, vst);
  };

  // 0) conversions
  cvt_bf16<<<dim3((int)(M * D / 256)), blk, 0, stream>>>(X,   Xb, (int)(M * D));
  cvt_bf16<<<dim3((int)(M * D / 256)), blk, 0, stream>>>(ENC, Eb, (int)(M * D));
  cvtw(Wq1, Wq1t, D, D); cvtw(Wk1, Wk1t, D, D); cvtw(Wv1, Wv1t, D, D); cvtw(Wo1, Wo1t, D, D);
  cvtw(Wq2, Wq2t, D, D); cvtw(Wk2, Wk2t, D, D); cvtw(Wv2, Wv2t, D, D); cvtw(Wo2, Wo2t, D, D);
  cvtw(Wf1, Wf1t, D, F); cvtw(Wf2, Wf2t, F, D);

  // 1) self-attention
  gemm(Xb, Wq1t, bq1, nullptr, nullptr, Qb,  M, D, D, 0, 0);
  gemm(Xb, Wk1t, bk1, nullptr, nullptr, Kb,  M, D, D, 0, 0);
  gemm(Xb, Wv1t, bv1, nullptr, nullptr, Vtb, M, D, D, 0, 1);  // transposed store
  attn_wmma<<<dim3(512), blk, 0, stream>>>(Qb, Kb, Vtb, Atn, VL, 1);
  gemm(Atn, Wo1t, bo1, X, Tmp, nullptr, M, D, D, 0, 0);       // + residual X
  ln_kernel<<<dim3((int)M), blk, 0, stream>>>(Tmp, g1, be1, Yf, Yb, (int)D);

  // 2) cross-attention
  gemm(Yb, Wq2t, bq2, nullptr, nullptr, Qb,  M, D, D, 0, 0);
  gemm(Eb, Wk2t, bk2, nullptr, nullptr, Kb,  M, D, D, 0, 0);
  gemm(Eb, Wv2t, bv2, nullptr, nullptr, Vtb, M, D, D, 0, 1);
  attn_wmma<<<dim3(512), blk, 0, stream>>>(Qb, Kb, Vtb, Atn, VL, 0);
  gemm(Atn, Wo2t, bo2, Yf, Tmp, nullptr, M, D, D, 0, 0);      // + residual Y
  ln_kernel<<<dim3((int)M), blk, 0, stream>>>(Tmp, g2, be2, Zf, Zb, (int)D);

  // 3) FFN
  gemm(Zb, Wf1t, bf1, nullptr, nullptr, Hb, M, F, D, 1, 0);   // ReLU, bf16 out
  gemm(Hb, Wf2t, bf2, Zf, Tmp, nullptr, M, D, F, 0, 0);       // + residual Z
  ln_kernel<<<dim3((int)M), blk, 0, stream>>>(Tmp, g3, be3, OUT, nullptr, (int)D);
}
